// AttentionBlock_17918603559502
// MI455X (gfx1250) — matrix operand order repaired and checked by reading, not compiled, not run on hardware
//
#include <hip/hip_runtime.h>
#include <hip/hip_bf16.h>

typedef __attribute__((ext_vector_type(16))) _Float16 v16h;
typedef __attribute__((ext_vector_type(8)))  float    v8f;
typedef int g_v4i __attribute__((vector_size(16)));

#define B_   16
#define C_   512
#define T_   2048
#define ROWS 640   // 64 (q) + 64 (k) + 512 (v)

union HF16 { uint4 u[2]; v16h v; };
union F4   { float4 f; float x[4]; };

// ---------------- gfx1250 async-to-LDS helpers (guarded) ----------------
#if defined(__AMDGCN__) && __has_builtin(__builtin_amdgcn_global_load_async_to_lds_b128)
#define HAVE_ASYNC 1
#define ASYNC_CP16(ldst, gsrc)                                                    \
    __builtin_amdgcn_global_load_async_to_lds_b128(                               \
        (__attribute__((address_space(1))) g_v4i*)(size_t)(const void*)(gsrc),    \
        (__attribute__((address_space(3))) g_v4i*)(void*)(ldst), 0, 0)
#if __has_builtin(__builtin_amdgcn_s_wait_asynccnt)
#define WAIT_ASYNC(n) __builtin_amdgcn_s_wait_asynccnt(n)
#else
#define WAIT_ASYNC(n) asm volatile("s_wait_asynccnt %0" ::"i"(n))
#endif
#else
#define HAVE_ASYNC 0
#define ASYNC_CP16(ldst, gsrc)                                                    \
    (*reinterpret_cast<uint4*>(ldst) = *reinterpret_cast<const uint4*>(gsrc))
#define WAIT_ASYNC(n)
#endif

// ---------------- Kernel 0: pack fused weights (fp32 -> f16) + bias ----------------
__global__ void pack_w(const float* __restrict__ wq, const float* __restrict__ bq,
                       const float* __restrict__ wk, const float* __restrict__ bk,
                       const float* __restrict__ wv, const float* __restrict__ bv,
                       _Float16* __restrict__ Wh, float* __restrict__ biasArr) {
    int idx = blockIdx.x * blockDim.x + threadIdx.x;
    if (idx < ROWS * C_) {
        int r = idx / C_, c = idx % C_;
        float v;
        if      (r < 64)  v = wq[r * C_ + c];
        else if (r < 128) v = wk[(r - 64) * C_ + c];
        else              v = wv[(r - 128) * C_ + c];
        Wh[idx] = (_Float16)v;
    }
    if (idx < ROWS) {
        float bb;
        if      (idx < 64)  bb = bq[idx];
        else if (idx < 128) bb = bk[idx - 64];
        else                bb = bv[idx - 128];
        biasArr[idx] = bb;
    }
}

// ---------------- Kernel 1: one-shot transpose x (f32 [c][t]) -> xh (f16 [t][c]) ----------------
#define TSTR 72
__global__ __launch_bounds__(256)
void xpose(const float* __restrict__ x, _Float16* __restrict__ xh) {
    __shared__ _Float16 tile[64 * TSTR];
    const int tid = threadIdx.x;
    const int t0 = blockIdx.x * 64, c0 = blockIdx.y * 64, b = blockIdx.z;
    {
        int c    = tid >> 2;            // 0..63
        int toff = (tid & 3) * 16;      // 0..48
        const float* src = &x[((size_t)b * C_ + c0 + c) * T_ + t0 + toff];
#pragma unroll
        for (int q = 0; q < 4; ++q) {
            F4 f; f.f = *reinterpret_cast<const float4*>(src + q * 4);
#pragma unroll
            for (int e = 0; e < 4; ++e)
                tile[(toff + q * 4 + e) * TSTR + c] = (_Float16)f.x[e];
        }
    }
    __syncthreads();
    {
        int t    = tid >> 2;
        int coff = (tid & 3) * 16;
        _Float16* dst = &xh[((size_t)b * T_ + t0 + t) * C_ + c0 + coff];
        *reinterpret_cast<uint4*>(dst)     = *reinterpret_cast<const uint4*>(&tile[t * TSTR + coff]);
        *reinterpret_cast<uint4*>(dst + 8) = *reinterpret_cast<const uint4*>(&tile[t * TSTR + coff + 8]);
    }
}

// ---------------- Kernel 2: QKV GEMM (640x512)x(512x2048)+bias, async double-buffered ----------------
#define WSTR 40   // 32 k halves + 8 pad
__global__ __launch_bounds__(256)
void qkv_gemm(const _Float16* __restrict__ xh, const _Float16* __restrict__ Wh,
              const float* __restrict__ biasArr, _Float16* __restrict__ qkvh,
              _Float16* __restrict__ qkT) {
    __shared__ _Float16 Wt[2][128 * WSTR];   // [row][k]
    __shared__ _Float16 Xt[2][128 * WSTR];   // [t][k]

    const int tid  = threadIdx.x;
    const int lane = tid & 31, g = lane >> 4, n = lane & 15;
    const int w    = tid >> 5;
    const int wr   = w >> 2, wc = w & 3;     // wave tile: 64 rows x 32 t
    const int r0   = blockIdx.x * 128;
    const int t0   = blockIdx.y * 128;
    const int b    = blockIdx.z;

    // one stage = 4 async b128 instructions per wave
    auto stage = [&](int kk, int buf) {
        int row = tid >> 1;
        int kh  = (tid & 1) * 16;
        const _Float16* wsrc = &Wh[(size_t)(r0 + row) * C_ + kk * 32 + kh];
        ASYNC_CP16(&Wt[buf][row * WSTR + kh],     wsrc);
        ASYNC_CP16(&Wt[buf][row * WSTR + kh + 8], wsrc + 8);
        const _Float16* xsrc = &xh[((size_t)b * T_ + t0 + row) * C_ + kk * 32 + kh];
        ASYNC_CP16(&Xt[buf][row * WSTR + kh],     xsrc);
        ASYNC_CP16(&Xt[buf][row * WSTR + kh + 8], xsrc + 8);
    };

    v8f acc[4][2] = {};
    stage(0, 0);
    stage(1, 1);

    for (int kk = 0; kk < 16; ++kk) {
        const int cur = kk & 1;
        if (kk < 15) { WAIT_ASYNC(4); } else { WAIT_ASYNC(0); }
        __syncthreads();

        v16h A[4], Bm[2];
#pragma unroll
        for (int i = 0; i < 4; ++i) {
            int row = wr * 64 + i * 16 + n;
            HF16 fa;
            fa.u[0] = *reinterpret_cast<const uint4*>(&Wt[cur][row * WSTR + 8 * g]);
            fa.u[1] = *reinterpret_cast<const uint4*>(&Wt[cur][row * WSTR + 16 + 8 * g]);
            A[i] = fa.v;
        }
#pragma unroll
        for (int j = 0; j < 2; ++j) {
            int tt = wc * 32 + j * 16 + n;
            HF16 fb;
            fb.u[0] = *reinterpret_cast<const uint4*>(&Xt[cur][tt * WSTR + 8 * g]);
            fb.u[1] = *reinterpret_cast<const uint4*>(&Xt[cur][tt * WSTR + 16 + 8 * g]);
            Bm[j] = fb.v;
        }
#pragma unroll
        for (int i = 0; i < 4; ++i)
#pragma unroll
            for (int j = 0; j < 2; ++j)
                acc[i][j] = __builtin_amdgcn_wmma_f32_16x16x32_f16(
                    false, A[i], false, Bm[j], (short)0, acc[i][j], false, false);
        __syncthreads();
        if (kk + 2 < 16) stage(kk + 2, cur);
    }

    // bias + store: Q/K rows go transposed to qkT[b][qk][t][d]; V rows to qkvh
#pragma unroll
    for (int i = 0; i < 4; ++i)
#pragma unroll
        for (int j = 0; j < 2; ++j) {
            int tt = t0 + wc * 32 + j * 16 + n;
#pragma unroll
            for (int v = 0; v < 8; ++v) {
                int row = r0 + wr * 64 + i * 16 + v + 8 * g;
                float val = acc[i][j][v] + biasArr[row];
                if (row < 128) {
                    int qk = row >> 6, dd = row & 63;
                    qkT[(((size_t)b * 2 + qk) * T_ + tt) * 64 + dd] = (_Float16)val;
                } else {
                    qkvh[((size_t)b * ROWS + row) * T_ + tt] = (_Float16)val;
                }
            }
        }
}

// ---------------- Kernel 3: flash attention + residual ----------------
#define DSTR 72   // 64 d halves + 8 pad
#define SSTR 68
#define PSTR 72
__global__ __launch_bounds__(512)
void attn(const _Float16* __restrict__ qkvh, const _Float16* __restrict__ qkT,
          const float* __restrict__ x, const float* __restrict__ gamma,
          float* __restrict__ out) {
    __shared__ _Float16 Qt[64 * DSTR];       // [t][d]
    __shared__ _Float16 Kt[2][64 * DSTR];    // [s][d], double buffered
    __shared__ float    Sf[64 * SSTR];       // [t][s] raw scores
    __shared__ _Float16 Sp[64 * PSTR];       // [t][s] exp probs (f16)
    __shared__ float mrow[64], lrow[64], arow[64];

    const int tid  = threadIdx.x;
    const int lane = tid & 31, g = lane >> 4, n = lane & 15;
    const int w    = tid >> 5;               // 0..15
    const int t0   = blockIdx.x * 64;
    const int b    = blockIdx.y;
    const size_t qbase = (size_t)b * ROWS * T_;
    const size_t qTb   = (size_t)b * 2 * T_ * 64;

    auto stageK = [&](int s0, int buf) {     // 1 async b128 instruction per wave
        int t = tid >> 3, doff = (tid & 7) * 8;
        ASYNC_CP16(&Kt[buf][t * DSTR + doff],
                   &qkT[qTb + (size_t)(T_ + s0 + t) * 64 + doff]);
    };
    {   // Q tile: straight copy (pre-transposed in workspace)
        int t = tid >> 3, doff = (tid & 7) * 8;
        ASYNC_CP16(&Qt[t * DSTR + doff], &qkT[qTb + (size_t)(t0 + t) * 64 + doff]);
    }
    stageK(0, 0);
    if (tid < 64) { mrow[tid] = -INFINITY; lrow[tid] = 0.f; }

    v8f acc[2][4] = {};                      // 32 ch x 64 t per wave
    const int ch0 = w * 32;

    for (int it = 0; it < T_ / 64; ++it) {
        const int s0 = it * 64, cur = it & 1;
        if (it + 1 < T_ / 64) { stageK(s0 + 64, 1 - cur); WAIT_ASYNC(1); }
        else                  { WAIT_ASYNC(0); }
        __syncthreads();

        // S = Q^T K : each wave one 16x16 tile (K-dim = 64)
        {
            const int ti = w >> 2, si = w & 3;
            v8f s = {};
#pragma unroll
            for (int kk = 0; kk < 2; ++kk) {
                HF16 fa, fb;
                int qr = (ti * 16 + n) * DSTR + kk * 32;
                fa.u[0] = *reinterpret_cast<const uint4*>(&Qt[qr + 8 * g]);
                fa.u[1] = *reinterpret_cast<const uint4*>(&Qt[qr + 16 + 8 * g]);
                int kr = (si * 16 + n) * DSTR + kk * 32;
                fb.u[0] = *reinterpret_cast<const uint4*>(&Kt[cur][kr + 8 * g]);
                fb.u[1] = *reinterpret_cast<const uint4*>(&Kt[cur][kr + 16 + 8 * g]);
                s = __builtin_amdgcn_wmma_f32_16x16x32_f16(
                    false, fa.v, false, fb.v, (short)0, s, false, false);
            }
#pragma unroll
            for (int v = 0; v < 8; ++v)
                Sf[(ti * 16 + v + 8 * g) * SSTR + si * 16 + n] = s[v];
        }
        __syncthreads();

        // online softmax: 8 threads per row
        {
            int t = tid >> 3, soff = (tid & 7) * 8;
            float mprev = mrow[t], lprev = lrow[t];
            float vals[8], mx = -INFINITY;
#pragma unroll
            for (int e = 0; e < 8; ++e) { vals[e] = Sf[t * SSTR + soff + e]; mx = fmaxf(mx, vals[e]); }
#pragma unroll
            for (int d = 1; d < 8; d <<= 1) mx = fmaxf(mx, __shfl_xor(mx, d, 8));
            float mnew  = fmaxf(mprev, mx);
            float alpha = __expf(mprev - mnew);
            float sum = 0.f;
#pragma unroll
            for (int e = 0; e < 8; ++e) {
                float p = __expf(vals[e] - mnew);
                sum += p;
                Sp[t * PSTR + soff + e] = (_Float16)p;
            }
#pragma unroll
            for (int d = 1; d < 8; d <<= 1) sum += __shfl_xor(sum, d, 8);
            if ((tid & 7) == 0) { mrow[t] = mnew; lrow[t] = alpha * lprev + sum; arow[t] = alpha; }
        }
        __syncthreads();

        // O update: rescale accumulator, then O += V_tile x P^T
        {
#pragma unroll
            for (int j = 0; j < 4; ++j) {
                float a = arow[j * 16 + n];
#pragma unroll
                for (int i = 0; i < 2; ++i)
#pragma unroll
                    for (int v = 0; v < 8; ++v) acc[i][j][v] *= a;
            }
#pragma unroll
            for (int kk = 0; kk < 2; ++kk) {
                v16h Va[2];
#pragma unroll
                for (int i = 0; i < 2; ++i) {
                    HF16 fa;
                    const _Float16* p =
                        &qkvh[qbase + (size_t)(128 + ch0 + i * 16 + n) * T_ + s0 + kk * 32];
                    fa.u[0] = *reinterpret_cast<const uint4*>(p + 8 * g);
                    fa.u[1] = *reinterpret_cast<const uint4*>(p + 16 + 8 * g);
                    Va[i] = fa.v;
                }
#pragma unroll
                for (int j = 0; j < 4; ++j) {
                    HF16 fb;
                    int pr = (j * 16 + n) * PSTR + kk * 32;
                    fb.u[0] = *reinterpret_cast<const uint4*>(&Sp[pr + 8 * g]);
                    fb.u[1] = *reinterpret_cast<const uint4*>(&Sp[pr + 16 + 8 * g]);
#pragma unroll
                    for (int i = 0; i < 2; ++i)
                        acc[i][j] = __builtin_amdgcn_wmma_f32_16x16x32_f16(
                            false, Va[i], false, fb.v, (short)0, acc[i][j], false, false);
                }
            }
        }
        __syncthreads();
    }

    // epilogue: normalize, gamma, residual
    float gf = gamma[0];
#pragma unroll
    for (int j = 0; j < 4; ++j) {
        float linv = 1.f / lrow[j * 16 + n];
        int tt = t0 + j * 16 + n;
#pragma unroll
        for (int i = 0; i < 2; ++i)
#pragma unroll
            for (int v = 0; v < 8; ++v) {
                int ch = ch0 + i * 16 + v + 8 * g;
                size_t idx = ((size_t)b * C_ + ch) * T_ + tt;
                out[idx] = gf * (acc[i][j][v] * linv) + x[idx];
            }
    }
}

extern "C" void kernel_launch(void* const* d_in, const int* in_sizes, int n_in,
                              void* d_out, int out_size, void* d_ws, size_t ws_size,
                              hipStream_t stream) {
    const float* x     = (const float*)d_in[0];
    const float* wq    = (const float*)d_in[1];
    const float* bq    = (const float*)d_in[2];
    const float* wk    = (const float*)d_in[3];
    const float* bk    = (const float*)d_in[4];
    const float* wv    = (const float*)d_in[5];
    const float* bv    = (const float*)d_in[6];
    const float* gamma = (const float*)d_in[7];
    float* out = (float*)d_out;

    char* ws = (char*)d_ws;
    size_t off = 0;
    _Float16* Wh      = (_Float16*)(ws + off); off += (size_t)ROWS * C_ * 2;        // 640 KB
    float*    biasArr = (float*)   (ws + off); off += (size_t)ROWS * 4;             // 2.5 KB
    _Float16* qkvh    = (_Float16*)(ws + off); off += (size_t)B_ * ROWS * T_ * 2;   // 40 MB
    _Float16* qkT     = (_Float16*)(ws + off); off += (size_t)B_ * 2 * T_ * 64 * 2; // 8 MB
    _Float16* xh      = (_Float16*)(ws + off); off += (size_t)B_ * T_ * C_ * 2;     // 32 MB

    pack_w<<<(ROWS * C_ + 255) / 256, 256, 0, stream>>>(wq, bq, wk, bk, wv, bv, Wh, biasArr);
    xpose<<<dim3(T_ / 64, C_ / 64, B_), 256, 0, stream>>>(x, xh);
    qkv_gemm<<<dim3(ROWS / 128, T_ / 128, B_), 256, 0, stream>>>(xh, Wh, biasArr, qkvh, qkT);
    attn<<<dim3(T_ / 64, B_), 512, 0, stream>>>(qkvh, qkT, x, gamma, out);
}
